// VectorQuantizer_34591666602323
// MI455X (gfx1250) — matrix operand order repaired and checked by reading, not compiled, not run on hardware
//
#include <hip/hip_runtime.h>

// ---------------------------------------------------------------------------
// VQ codebook lookup on gfx1250 (MI455X).
//   D^T = E(1024x64) x z^T(64xN)  via v_wmma_f32_16x16x32_f16, argmin, gather.
// z is channel-major, which IS the WMMA B-operand layout (K=channel striped
// over VGPRs, N=spatial striped over lanes) -> zero transposes anywhere.
// Embedding scaled by 1024 into f16 (argmin invariant under scaling).
// ---------------------------------------------------------------------------

typedef __attribute__((ext_vector_type(16))) _Float16 v16h;
typedef __attribute__((ext_vector_type(8)))  _Float16 v8h;
typedef __attribute__((ext_vector_type(8)))  float    v8f;

#define EMBED_NUM 1024
#define EMBED_DIM 64
#define SPATIAL   32768            // 32*32*32
#define ESCALE    1024.0f

// --- prep: f32 codebook -> scaled f16 copy in d_ws, plus ||e'||^2 (f32) -----
__global__ __launch_bounds__(256) void vq_prep_kernel(
    const float* __restrict__ emb, _Float16* __restrict__ emb16,
    float* __restrict__ e2) {
  const int g = blockIdx.x * 256 + threadIdx.x;     // 0..16383
  const float4 v = reinterpret_cast<const float4*>(emb)[g];
  _Float16* p = emb16 + g * 4;
  p[0] = (_Float16)(v.x * ESCALE);
  p[1] = (_Float16)(v.y * ESCALE);
  p[2] = (_Float16)(v.z * ESCALE);
  p[3] = (_Float16)(v.w * ESCALE);
  if (g < EMBED_NUM) {
    const float* row = emb + g * EMBED_DIM;
    float s = 0.f;
#pragma unroll 8
    for (int c = 0; c < EMBED_DIM; ++c) {
      const float t = (float)(_Float16)(row[c] * ESCALE);
      s += t * t;
    }
    e2[g] = s;
  }
}

// --- main: per wave = 16 spatial positions vs all 1024 codes ----------------
__global__ __launch_bounds__(256) void vq_main_kernel(
    const float*    __restrict__ z,
    const float*    __restrict__ embf,
    const _Float16* __restrict__ emb16,
    const float*    __restrict__ e2,
    float*          __restrict__ out) {
  const int wave = threadIdx.x >> 5;
  const int lane = threadIdx.x & 31;
  const int n    = lane & 15;                 // spatial column within tile
  const bool hi  = lane >= 16;                // upper lane half

  const int gt = blockIdx.x * 8 + wave;       // global 16-position tile id
  const int sg = gt << 4;                     // flat spatial start
  const int b  = sg >> 15;                    // batch (SPATIAL = 2^15)
  const int s  = sg & (SPATIAL - 1);          // spatial start within batch

  const float* zb = z + (size_t)b * EMBED_DIM * SPATIAL + s + n;

  // ---- WMMA B operands: z tile (64 ch x 16 spatial), f32 -> f16 -----------
  // B layout (16-bit, 32x16): lanes 0-15 hold K=0..7 then 16..23 (N=lane);
  // lanes 16-31 hold K=8..15 then 24..31 (N=lane-16).
  union { v16h v; _Float16 h[16]; } B1, B2;
  const int cb = hi ? 8 : 0;
#pragma unroll
  for (int vv = 0; vv < 8; ++vv) {
    const int c0 = cb + 2 * vv + (vv >= 4 ? 8 : 0);
    B1.h[2*vv]   = (_Float16)zb[(size_t)(c0     ) * SPATIAL];
    B1.h[2*vv+1] = (_Float16)zb[(size_t)(c0 +  1) * SPATIAL];
    B2.h[2*vv]   = (_Float16)zb[(size_t)(c0 + 32) * SPATIAL];
    B2.h[2*vv+1] = (_Float16)zb[(size_t)(c0 + 33) * SPATIAL];
  }

  float minv = 3.4e38f;
  int   mini = 0;
  const int koff = hi ? 8 : 0;                // A-operand per-lane K chunk

  for (int t = 0; t < 64; ++t) {
    // A operand: 16 embedding rows (M = lane&15), row-major f16 in L2.
    // Layout (16-bit, 16x32): lanes<16 hold K=0..7 then 16..23; lanes>=16
    // hold K=8..15 then 24..31.
    const _Float16* ar = emb16 + (size_t)(t * 16 + n) * EMBED_DIM;
    const v8h a1l = *(const v8h*)(ar + koff);
    const v8h a1h = *(const v8h*)(ar + 16 + koff);
    const v8h a2l = *(const v8h*)(ar + 32 + koff);
    const v8h a2h = *(const v8h*)(ar + 48 + koff);
    const v16h A1 = __builtin_shufflevector(a1l, a1h,
        0,1,2,3,4,5,6,7,8,9,10,11,12,13,14,15);
    const v16h A2 = __builtin_shufflevector(a2l, a2h,
        0,1,2,3,4,5,6,7,8,9,10,11,12,13,14,15);

    v8f acc = {};
    acc = __builtin_amdgcn_wmma_f32_16x16x32_f16(false, A1, false, B1.v,
                                                 (short)0, acc, false, false);
    acc = __builtin_amdgcn_wmma_f32_16x16x32_f16(false, A2, false, B2.v,
                                                 (short)0, acc, false, false);

    // D layout: lanes<16 get rows M=0..7 in acc[0..7]; lanes>=16 get M=8..15.
    const int code0 = t * 16 + (hi ? 8 : 0);
    const float4 q0 = *(const float4*)(e2 + code0);
    const float4 q1 = *(const float4*)(e2 + code0 + 4);
    float sc[8];
    sc[0] = q0.x - 2.f * acc[0];  sc[1] = q0.y - 2.f * acc[1];
    sc[2] = q0.z - 2.f * acc[2];  sc[3] = q0.w - 2.f * acc[3];
    sc[4] = q1.x - 2.f * acc[4];  sc[5] = q1.y - 2.f * acc[5];
    sc[6] = q1.z - 2.f * acc[6];  sc[7] = q1.w - 2.f * acc[7];
#pragma unroll
    for (int r = 0; r < 8; ++r)
      if (sc[r] < minv) { minv = sc[r]; mini = code0 + r; }
  }

  // lanes l and l^16 hold disjoint code subsets for the same spatial column
  {
    const float ov = __shfl_xor(minv, 16);
    const int   oi = __shfl_xor(mini, 16);
    if (ov < minv || (ov == minv && oi < mini)) { minv = ov; mini = oi; }
  }

  // ---- gather the winning f32 codebook row, channel-split coalesced store --
  const int c2 = hi ? 32 : 0;                 // lanes<16: ch 0..31, else 32..63
  const float* er = embf + (size_t)mini * EMBED_DIM + c2;
  float* ob = out + (size_t)b * EMBED_DIM * SPATIAL + s + n;
#pragma unroll 8
  for (int c = 0; c < 32; ++c)
    ob[(size_t)(c2 + c) * SPATIAL] = er[c];
}

extern "C" void kernel_launch(void* const* d_in, const int* in_sizes, int n_in,
                              void* d_out, int out_size, void* d_ws, size_t ws_size,
                              hipStream_t stream) {
  const float* z   = (const float*)d_in[0];   // (4, 64, 32, 32, 32) f32
  const float* emb = (const float*)d_in[1];   // (1024, 64) f32
  float* out = (float*)d_out;                 // (4, 64, 32, 32, 32) f32

  _Float16* emb16 = (_Float16*)d_ws;                                  // 128 KB
  float*    e2    = (float*)((char*)d_ws +
                             (size_t)EMBED_NUM * EMBED_DIM * sizeof(_Float16)); // 4 KB

  vq_prep_kernel<<<dim3(64), dim3(256), 0, stream>>>(emb, emb16, e2);
  // 131072 spatial positions / 16 per wave / 8 waves per block = 1024 blocks
  vq_main_kernel<<<dim3(1024), dim3(256), 0, stream>>>(z, emb, emb16, e2, out);
}
